// SelfAttention_74990128988510
// MI455X (gfx1250) — compile-verified, matrix operand order checked by reading, not run on hardware
//
#include <hip/hip_runtime.h>

#define D_EMB 1024
#define SEQ   2048
#define BATCH 4

typedef __attribute__((ext_vector_type(16))) __bf16       v16bf;
typedef __attribute__((ext_vector_type(8)))  float        v8f;
typedef __attribute__((ext_vector_type(4)))  unsigned int v4u;

// ---------- scalar helpers ----------
__device__ __forceinline__ unsigned short f2bf(float f) {
  union { float f; unsigned int u; } v; v.f = f;
  unsigned int u = v.u;
  u += 0x7FFFu + ((u >> 16) & 1u);       // round-to-nearest-even
  return (unsigned short)(u >> 16);
}
__device__ __forceinline__ float bf2f(unsigned short h) {
  union { unsigned int u; float f; } v; v.u = ((unsigned int)h) << 16;
  return v.f;
}

// ---------- WMMA fragment load ----------
// 16-bit A fragment (16x32 MxK) from row-major MxK memory; the B fragment
// (32x16 KxN) has the mirrored per-lane pattern, so it loads identically from
// an NxK row-major matrix. Each lane: two contiguous 16-byte loads.
__device__ __forceinline__ v16bf load_frag(const unsigned short* base, int ld, int lane) {
  const int r  = lane & 15;            // row (A) / column (B)
  const int k0 = (lane >> 4) << 3;     // lanes 16-31 hold K offset +8
  union { v4u u[2]; v16bf f; } t;
  t.u[0] = *(const v4u*)(base + r * ld + k0);        // K = k0 .. k0+7
  t.u[1] = *(const v4u*)(base + r * ld + k0 + 16);   // K = k0+16 .. k0+23
  return t.f;
}

__device__ __forceinline__ v8f wmma_bf16(v16bf a, v16bf b, v8f c) {
  return __builtin_amdgcn_wmma_f32_16x16x32_bf16(
      /*neg_a=*/false, a, /*neg_b=*/false, b,
      /*c_mod=*/(short)0, c, /*reuse_a=*/false, /*reuse_b=*/false);
}

// ---------- wave-level 64x64 GEMM accumulation ----------
// acc[4][4] covers M = 4x16, N = 4x16.  A: MxK row-major.  Bt: NxK row-major.
// 16 b128 loads feed 16 WMMAs per K=32 step (1.0 loads per WMMA).
__device__ __forceinline__ void gemm_wave_64x64(
    const unsigned short* __restrict__ A, int lda,
    const unsigned short* __restrict__ Bt, int ldb,
    int K, int lane, v8f acc[4][4]) {
  for (int k0 = 0; k0 < K; k0 += 32) {
    v16bf a[4], b[4];
    #pragma unroll
    for (int i = 0; i < 4; ++i)
      a[i] = load_frag(A + (size_t)(16 * i) * lda + k0, lda, lane);
    #pragma unroll
    for (int j = 0; j < 4; ++j)
      b[j] = load_frag(Bt + (size_t)(16 * j) * ldb + k0, ldb, lane);
    #pragma unroll
    for (int i = 0; i < 4; ++i)
      #pragma unroll
      for (int j = 0; j < 4; ++j)
        acc[i][j] = wmma_bf16(a[i], b[j], acc[i][j]);
  }
}

// ---------- kernel 1: fp32 -> bf16 conversion of x (4 elems / thread) ----------
__global__ void __launch_bounds__(256)
cvt_x_kernel(const float* __restrict__ x, unsigned short* __restrict__ xb) {
  const size_t i = ((size_t)blockIdx.x * 256 + threadIdx.x) * 4;
  const float f0 = x[i + 0], f1 = x[i + 1], f2 = x[i + 2], f3 = x[i + 3];
  unsigned long long packed =
      (unsigned long long)f2bf(f0)
    | ((unsigned long long)f2bf(f1) << 16)
    | ((unsigned long long)f2bf(f2) << 32)
    | ((unsigned long long)f2bf(f3) << 48);
  *(unsigned long long*)(xb + i) = packed;
}

// ---------- kernel 2: transpose W[k][n] -> Wt[n][k] in bf16 (LDS tiled) ----------
__global__ void __launch_bounds__(256)
wt_kernel(const float* __restrict__ Wq, const float* __restrict__ Wk,
          const float* __restrict__ Wv, unsigned short* __restrict__ wt) {
  __shared__ float tile[32][33];
  const int z = blockIdx.z;
  const float* W = (z == 0) ? Wq : (z == 1) ? Wk : Wv;
  unsigned short* dst = wt + (size_t)z * D_EMB * D_EMB;
  const int bx = blockIdx.x * 32;      // n
  const int by = blockIdx.y * 32;      // k
  const int tx = threadIdx.x & 31;
  const int ty = threadIdx.x >> 5;     // 0..7
  #pragma unroll
  for (int j = ty; j < 32; j += 8)
    tile[j][tx] = W[(size_t)(by + j) * D_EMB + bx + tx];
  __syncthreads();
  #pragma unroll
  for (int j = ty; j < 32; j += 8)
    dst[(size_t)(bx + j) * D_EMB + by + tx] = f2bf(tile[tx][j]);
}

// ---------- kernel 3: QKV projections (z selects Q/K/V) ----------
// block tile 128(M) x 256(N); 8 waves arranged 2(M) x 4(N), each 64x64.
__global__ void __launch_bounds__(256)
qkv_gemm_kernel(const unsigned short* __restrict__ xb,   // [8192][1024] bf16
                const unsigned short* __restrict__ wt,   // [3][1024][1024] NxK bf16
                const float* __restrict__ bq,
                const float* __restrict__ bk,
                const float* __restrict__ bv,
                unsigned short* __restrict__ Qb,   // [8192][1024] (pre-scaled 1/32)
                unsigned short* __restrict__ Kb,   // [8192][1024]
                unsigned short* __restrict__ Vt) { // [4][1024][2048] transposed
  const int lane = threadIdx.x & 31;
  const int wave = threadIdx.x >> 5;
  const int z  = blockIdx.z;
  const int m0 = blockIdx.x * 128 + (wave >> 2) * 64;
  const int n0 = blockIdx.y * 256 + (wave & 3) * 64;

  const unsigned short* A  = xb + (size_t)m0 * D_EMB;
  const unsigned short* Bt = wt + (size_t)z * D_EMB * D_EMB + (size_t)n0 * D_EMB;

  v8f acc[4][4] = {};
  gemm_wave_64x64(A, D_EMB, Bt, D_EMB, D_EMB, lane, acc);

  const float* bias = (z == 0) ? bq : (z == 1) ? bk : bv;
  const float scale = (z == 0) ? 0.03125f : 1.0f;   // fold 1/sqrt(1024) into Q
  const int nLane = lane & 15;
  const int mOff  = (lane >> 4) * 8;
  #pragma unroll
  for (int i = 0; i < 4; ++i) {
    #pragma unroll
    for (int j = 0; j < 4; ++j) {
      const int nG = n0 + j * 16 + nLane;
      const float b = bias[nG];
      #pragma unroll
      for (int e = 0; e < 8; ++e) {
        const int mG = m0 + i * 16 + mOff + e;
        const unsigned short h = f2bf((acc[i][j][e] + b) * scale);
        if (z == 0) {
          Qb[(size_t)mG * D_EMB + nG] = h;
        } else if (z == 1) {
          Kb[(size_t)mG * D_EMB + nG] = h;
        } else {
          const int bb = mG >> 11;
          const int s  = mG & (SEQ - 1);
          Vt[((size_t)bb * D_EMB + nG) * SEQ + s] = h;
        }
      }
    }
  }
}

// ---------- kernel 4: scores = (Q/32) @ K^T, bf16 out ----------
__global__ void __launch_bounds__(256)
scores_gemm_kernel(const unsigned short* __restrict__ Qb,
                   const unsigned short* __restrict__ Kb,
                   unsigned short* __restrict__ S) {   // [4][2048][2048]
  const int lane = threadIdx.x & 31;
  const int wave = threadIdx.x >> 5;
  const int b  = blockIdx.z;
  const int m0 = blockIdx.x * 128 + (wave >> 2) * 64;
  const int n0 = blockIdx.y * 256 + (wave & 3) * 64;

  const unsigned short* A  = Qb + ((size_t)b * SEQ + m0) * D_EMB;
  const unsigned short* Bt = Kb + ((size_t)b * SEQ + n0) * D_EMB;

  v8f acc[4][4] = {};
  gemm_wave_64x64(A, D_EMB, Bt, D_EMB, D_EMB, lane, acc);

  unsigned short* out = S + (size_t)b * SEQ * SEQ;
  const int nLane = lane & 15;
  const int mOff  = (lane >> 4) * 8;
  #pragma unroll
  for (int i = 0; i < 4; ++i)
    #pragma unroll
    for (int j = 0; j < 4; ++j)
      #pragma unroll
      for (int e = 0; e < 8; ++e)
        out[(size_t)(m0 + i * 16 + mOff + e) * SEQ + n0 + j * 16 + nLane] =
            f2bf(acc[i][j][e]);
}

// ---------- kernel 5: in-place row softmax over bf16 scores ----------
__global__ void __launch_bounds__(256)
softmax_kernel(unsigned short* __restrict__ S) {
  __shared__ float red[256];
  unsigned short* p = S + (size_t)blockIdx.x * SEQ;
  const int t = threadIdx.x;

  float v[8];
  float m = -3.0e38f;
  #pragma unroll
  for (int j = 0; j < 8; ++j) {
    v[j] = bf2f(p[t + j * 256]);
    m = fmaxf(m, v[j]);
  }
  red[t] = m;
  __syncthreads();
  for (int off = 128; off > 0; off >>= 1) {
    if (t < off) red[t] = fmaxf(red[t], red[t + off]);
    __syncthreads();
  }
  m = red[0];
  __syncthreads();

  float s = 0.0f;
  #pragma unroll
  for (int j = 0; j < 8; ++j) {
    v[j] = __expf(v[j] - m);
    s += v[j];
  }
  red[t] = s;
  __syncthreads();
  for (int off = 128; off > 0; off >>= 1) {
    if (t < off) red[t] += red[t + off];
    __syncthreads();
  }
  const float inv = 1.0f / red[0];
  #pragma unroll
  for (int j = 0; j < 8; ++j)
    p[t + j * 256] = f2bf(v[j] * inv);
}

// ---------- kernel 6: out = P @ V (B from transposed V), fp32 out ----------
__global__ void __launch_bounds__(256)
out_gemm_kernel(const unsigned short* __restrict__ P,    // [4][2048][2048]
                const unsigned short* __restrict__ Vt,   // [4][1024][2048]
                float* __restrict__ out) {               // [4][2048][1024]
  const int lane = threadIdx.x & 31;
  const int wave = threadIdx.x >> 5;
  const int b  = blockIdx.z;
  const int m0 = blockIdx.x * 128 + (wave >> 2) * 64;    // seq
  const int n0 = blockIdx.y * 256 + (wave & 3) * 64;     // feature

  const unsigned short* A  = P  + ((size_t)b * SEQ + m0) * SEQ;
  const unsigned short* Bt = Vt + ((size_t)b * D_EMB + n0) * SEQ;

  v8f acc[4][4] = {};
  gemm_wave_64x64(A, SEQ, Bt, SEQ, SEQ, lane, acc);

  const int nLane = lane & 15;
  const int mOff  = (lane >> 4) * 8;
  #pragma unroll
  for (int i = 0; i < 4; ++i)
    #pragma unroll
    for (int j = 0; j < 4; ++j)
      #pragma unroll
      for (int e = 0; e < 8; ++e)
        out[((size_t)b * SEQ + m0 + i * 16 + mOff + e) * D_EMB + n0 + j * 16 + nLane] =
            acc[i][j][e];
}

extern "C" void kernel_launch(void* const* d_in, const int* in_sizes, int n_in,
                              void* d_out, int out_size, void* d_ws, size_t ws_size,
                              hipStream_t stream) {
  const float* x  = (const float*)d_in[0];
  const float* Wq = (const float*)d_in[1];
  const float* bq = (const float*)d_in[2];
  const float* Wk = (const float*)d_in[3];
  const float* bk = (const float*)d_in[4];
  const float* Wv = (const float*)d_in[5];
  const float* bv = (const float*)d_in[6];
  float* out = (float*)d_out;

  char* ws = (char*)d_ws;
  unsigned short* xb = (unsigned short*)(ws);                          // 16 MB
  unsigned short* wt = (unsigned short*)(ws + (16ull << 20));          //  6 MB
  unsigned short* Qb = (unsigned short*)(ws + (22ull << 20));          // 16 MB
  unsigned short* Kb = (unsigned short*)(ws + (38ull << 20));          // 16 MB
  unsigned short* Vt = (unsigned short*)(ws + (54ull << 20));          // 16 MB
  unsigned short* S  = (unsigned short*)(ws + (70ull << 20));          // 32 MB

  cvt_x_kernel      <<<dim3(8192),       dim3(256), 0, stream>>>(x, xb);
  wt_kernel         <<<dim3(32, 32, 3),  dim3(256), 0, stream>>>(Wq, Wk, Wv, wt);
  qkv_gemm_kernel   <<<dim3(64, 4, 3),   dim3(256), 0, stream>>>(xb, wt, bq, bk, bv, Qb, Kb, Vt);
  scores_gemm_kernel<<<dim3(16, 8, 4),   dim3(256), 0, stream>>>(Qb, Kb, S);
  softmax_kernel    <<<dim3(8192),       dim3(256), 0, stream>>>(S);
  out_gemm_kernel   <<<dim3(16, 4, 4),   dim3(256), 0, stream>>>(S, Vt, out);
}
